// FlowAttention_82377472737910
// MI455X (gfx1250) — matrix operand order repaired and checked by reading, not compiled, not run on hardware
//
#include <hip/hip_runtime.h>
#include <stdint.h>

// ---------------------------------------------------------------------------
// FlowAttention for MI455X (gfx1250, wave32, WMMA).
//   B=4, S=8192, H=1024, NH=16, HD=64.  All heavy math in bf16 WMMA
//   (v_wmma_f32_16x16x32_bf16), fp32 accumulate.  GEMMs use the CDNA5
//   async global->LDS path (global_load_async_to_lds_b128, ASYNCcnt) with
//   double-buffered LDS so the tile copy for step k+1 overlaps WMMA on k.
// Workspace (~345 MB): Xb | Wqb Wkb Wvb Wob | Qm Km Vm | Attn | KV | Ksum
// ---------------------------------------------------------------------------

#define BATCH   4
#define SEQ     8192
#define HID     1024
#define NHEADS  16
#define HDIM    64
#define MTOT    (BATCH * SEQ)   // 32768

typedef __attribute__((ext_vector_type(16))) __bf16 v16bf;
typedef __attribute__((ext_vector_type(8)))  float  v8f;

union FragAB { v16bf v; uint4 q[2]; unsigned short h[16]; };
union FragC  { v8f   v; float f[8]; };

__device__ __forceinline__ float bf2f(unsigned short u) {
    return __uint_as_float(((unsigned int)u) << 16);
}
__device__ __forceinline__ unsigned short f2bf(float f) {
    unsigned int u = __float_as_uint(f);
    unsigned int r = u + 0x7FFFu + ((u >> 16) & 1u);   // round-to-nearest-even
    return (unsigned short)(r >> 16);
}

// LDS byte offset of a __shared__ object (addrspace(3) pointer value == offset)
__device__ __forceinline__ unsigned ldsAddr(const void* p) {
    return (unsigned)(unsigned long long)
        (const __attribute__((address_space(3))) char*)p;
}

// CDNA5 async copy: 32 bytes (global -> LDS) per lane, tracked by ASYNCcnt.
// INST_OFFSET is added on both the LDS and global side (ISA 08, GV mode).
__device__ __forceinline__ void asyncCopy32B(const unsigned short* g, unsigned lds) {
    asm volatile("global_load_async_to_lds_b128 %0, %1, off"
                 :: "v"(lds), "v"(g) : "memory");
    asm volatile("global_load_async_to_lds_b128 %0, %1, off offset:16"
                 :: "v"(lds), "v"(g) : "memory");
}

#if __has_builtin(__builtin_amdgcn_s_wait_asynccnt)
#define WAIT_ASYNC0() __builtin_amdgcn_s_wait_asynccnt(0)
#else
#define WAIT_ASYNC0() asm volatile("s_wait_asynccnt 0" ::: "memory")
#endif

// ---------------------------------------------------------------------------
__global__ void k_f32_to_bf16(const float* __restrict__ src,
                              unsigned short* __restrict__ dst, int n) {
    int i = blockIdx.x * blockDim.x + threadIdx.x;
    int stride = gridDim.x * blockDim.x;
    for (; i < n; i += stride) dst[i] = f2bf(src[i]);
}

__global__ void k_zero(float* __restrict__ p, int n) {
    int i = blockIdx.x * blockDim.x + threadIdx.x;
    if (i < n) p[i] = 0.0f;
}

// ---------------------------------------------------------------------------
// GEMM: C[m,n] = sum_k A[m,k] * W[n,k] + bias[n], fused epilogue.
// mode 0: V (identity, bf16 head-layout out)
// mode 1: Q (elu+1, bf16 head-layout out)
// mode 2: K ((elu+1)*exp(g)*mask, bf16 head-layout out)
// mode 3: O (identity, fp32 row-major out)
// Block 256 = 8 waves (4 in M x 2 in N).  BM=128, BN=128, BK=32.
// Wave tile 32x64: 8 WMMAs per K-step, fragments preloaded to overlap LDS.
// Double-buffered LDS filled by global_load_async_to_lds_b128.
// ---------------------------------------------------------------------------
__global__ __launch_bounds__(256)
void k_gemm(const unsigned short* __restrict__ A,
            const unsigned short* __restrict__ Wt,
            const float* __restrict__ bias,
            unsigned short* __restrict__ outB,
            float* __restrict__ outF,
            const float* __restrict__ gb,
            const int* __restrict__ mask,
            int mode)
{
    __shared__ __align__(16) unsigned short As[2][128][32];  // 2 x 8 KB
    __shared__ __align__(16) unsigned short Bs[2][128][32];  // 2 x 8 KB

    const int tid  = threadIdx.x;
    const int wave = tid >> 5;
    const int lane = tid & 31;
    const int lr   = lane & 15;
    const bool hi  = lane >= 16;
    const int wm = wave >> 1;        // 0..3 : M sub-tile (32 rows)
    const int wn = wave & 1;         // 0..1 : N sub-tile (64 cols)
    const int m0 = blockIdx.y * 128;
    const int n0 = blockIdx.x * 128;

    // Staging: each thread owns one 32B chunk of A-tile and one of B-tile.
    const int rowS = tid >> 1;             // 0..127
    const int colS = (tid & 1) * 16;       // ushort col 0 or 16

    FragC acc[2][4];
    #pragma unroll
    for (int i = 0; i < 2; ++i)
        #pragma unroll
        for (int j = 0; j < 4; ++j)
            #pragma unroll
            for (int r = 0; r < 8; ++r) acc[i][j].f[r] = 0.0f;

    // Prologue: fill buffer 0.
    asyncCopy32B(A  + (size_t)(m0 + rowS) * HID + colS, ldsAddr(&As[0][rowS][colS]));
    asyncCopy32B(Wt + (size_t)(n0 + rowS) * HID + colS, ldsAddr(&Bs[0][rowS][colS]));

    const int NT = HID / 32;   // 32 K-steps
    for (int kt = 0; kt < NT; ++kt) {
        const int buf = kt & 1;
        WAIT_ASYNC0();          // this wave's async writes to LDS landed
        __syncthreads();        // all waves' tiles ready; prev readers done

        if (kt + 1 < NT) {      // overlap: fill the other buffer
            const int k1 = (kt + 1) * 32;
            asyncCopy32B(A  + (size_t)(m0 + rowS) * HID + k1 + colS,
                         ldsAddr(&As[buf ^ 1][rowS][colS]));
            asyncCopy32B(Wt + (size_t)(n0 + rowS) * HID + k1 + colS,
                         ldsAddr(&Bs[buf ^ 1][rowS][colS]));
        }

        // Preload all fragments (distinct regs -> overlapped LDS latency).
        // A frag: lane<16 row lr K{0..7,16..23}; lane>=16 K{8..15,24..31}.
        FragAB a[2], b[4];
        #pragma unroll
        for (int i = 0; i < 2; ++i) {
            const int ar = wm * 32 + i * 16 + lr;
            a[i].q[0] = *(const uint4*)&As[buf][ar][hi ? 8 : 0];
            a[i].q[1] = *(const uint4*)&As[buf][ar][(hi ? 8 : 0) + 16];
        }
        #pragma unroll
        for (int j = 0; j < 4; ++j) {
            const int br = wn * 64 + j * 16 + lr;   // B: lane n = column n over K
            const int bc = hi ? 8 : 0;
            b[j].q[0] = *(const uint4*)&Bs[buf][br][bc];
            b[j].q[1] = *(const uint4*)&Bs[buf][br][bc + 16];
        }
        #pragma unroll
        for (int i = 0; i < 2; ++i)
            #pragma unroll
            for (int j = 0; j < 4; ++j)
                acc[i][j].v = __builtin_amdgcn_wmma_f32_16x16x32_bf16(
                    false, a[i].v, false, b[j].v, (short)0, acc[i][j].v, false, false);
    }

    // Epilogue. C VGPR r: M = r (lane<16) / r+8 (lane>=16), N = lr.
    #pragma unroll
    for (int i = 0; i < 2; ++i) {
        #pragma unroll
        for (int j = 0; j < 4; ++j) {
            #pragma unroll
            for (int r = 0; r < 8; ++r) {
                const int m = m0 + wm * 32 + i * 16 + (hi ? r + 8 : r);
                const int n = n0 + wn * 64 + j * 16 + lr;
                float y = acc[i][j].f[r] + bias[n];
                if (mode == 3) { outF[(size_t)m * HID + n] = y; continue; }
                float v;
                if (mode == 0) {
                    v = y;                                         // V
                } else {
                    float e = (y > 0.0f) ? (y + 1.0f) : __expf(y); // elu(y)+1
                    if (mode == 1) v = e;                          // Q
                    else {                                         // K
                        int bb = m >> 13, ss = m & (SEQ - 1);
                        v = mask[bb * SEQ + ss] ? e * __expf(gb[bb * SEQ + ss]) : 0.0f;
                    }
                }
                const int bb = m >> 13, ss = m & (SEQ - 1);
                const int h = n >> 6, d = n & 63;
                outB[((size_t)(bb * NHEADS + h) * SEQ + ss) * HDIM + d] = f2bf(v);
            }
        }
    }
}

// ---------------------------------------------------------------------------
// kv_context[g,d,v] = sum_s K[g,s,d]*V[g,s,v];  k_sum[g,d] = sum_s K[g,s,d]
// Grid (16 S-splits, 64 heads). K/V tiles transposed into LDS so the
// contraction (s) is contiguous per lane for both A and B fragments.
// ---------------------------------------------------------------------------
__global__ __launch_bounds__(256)
void k_kv(const unsigned short* __restrict__ Km,
          const unsigned short* __restrict__ Vm,
          float* __restrict__ KV, float* __restrict__ Ksum)
{
    __shared__ __align__(16) unsigned short Kt[64][32];   // [d][s_local]
    __shared__ __align__(16) unsigned short Vt[64][32];   // [v][s_local]

    const int tid  = threadIdx.x;
    const int wave = tid >> 5;
    const int lane = tid & 31;
    const int lr   = lane & 15;
    const bool hi  = lane >= 16;
    const int g  = blockIdx.y;
    const int s0 = blockIdx.x * 512;
    const int dt  = wave >> 1;              // d-tile 0..3
    const int vt0 = (wave & 1) * 2;         // v-tiles {0,1} or {2,3}

    FragC acc[2];
    #pragma unroll
    for (int i = 0; i < 2; ++i)
        #pragma unroll
        for (int r = 0; r < 8; ++r) acc[i].f[r] = 0.0f;
    float ksum = 0.0f;

    const int rowL = tid >> 3;              // s_local 0..31
    const int c8   = (tid & 7) * 8;         // d/v chunk base

    for (int it = 0; it < 16; ++it) {
        const int s = s0 + it * 32 + rowL;
        const size_t base = ((size_t)g * SEQ + s) * HDIM + c8;
        uint4 kvec = *(const uint4*)(Km + base);
        uint4 vvec = *(const uint4*)(Vm + base);
        __syncthreads();
        const unsigned short* kp = (const unsigned short*)&kvec;
        const unsigned short* vp = (const unsigned short*)&vvec;
        #pragma unroll
        for (int j = 0; j < 8; ++j) {       // transpose on store
            Kt[c8 + j][rowL] = kp[j];
            Vt[c8 + j][rowL] = vp[j];
        }
        __syncthreads();

        if (tid < 64) {
            #pragma unroll
            for (int sl = 0; sl < 32; ++sl) ksum += bf2f(Kt[tid][sl]);
        }

        FragAB a, b[2];                      // A[d,s] from Kt, B[s,v] from Vt
        a.q[0] = *(const uint4*)&Kt[dt * 16 + lr][hi ? 8 : 0];
        a.q[1] = *(const uint4*)&Kt[dt * 16 + lr][(hi ? 8 : 0) + 16];
        #pragma unroll
        for (int i = 0; i < 2; ++i) {
            const int bc = hi ? 8 : 0;
            b[i].q[0] = *(const uint4*)&Vt[(vt0 + i) * 16 + lr][bc];
            b[i].q[1] = *(const uint4*)&Vt[(vt0 + i) * 16 + lr][bc + 16];
        }
        #pragma unroll
        for (int i = 0; i < 2; ++i)
            acc[i].v = __builtin_amdgcn_wmma_f32_16x16x32_bf16(
                false, a.v, false, b[i].v, (short)0, acc[i].v, false, false);
    }

    if (tid < 64) atomicAdd(&Ksum[(size_t)g * 64 + tid], ksum);
    #pragma unroll
    for (int i = 0; i < 2; ++i) {
        #pragma unroll
        for (int r = 0; r < 8; ++r) {
            const int d = dt * 16 + (hi ? r + 8 : r);
            const int v = (vt0 + i) * 16 + lr;
            atomicAdd(&KV[((size_t)g * 64 + d) * 64 + v], acc[i].f[r]);
        }
    }
}

// ---------------------------------------------------------------------------
// attn[b,s,h*64+v] = (q[s,:]*scale) @ KV[g];  scale = nan_to_num(1/(q.ksum))
// Grid (64 S-blocks of 128 rows, 64 heads); 8 waves, 16 rows each.
// ---------------------------------------------------------------------------
__global__ __launch_bounds__(256)
void k_attn(const unsigned short* __restrict__ Qm,
            const float* __restrict__ KV,
            const float* __restrict__ Ksum,
            unsigned short* __restrict__ Attn)
{
    __shared__ __align__(16) unsigned short KVt[64][64];      // [v][d] bf16
    __shared__ __align__(16) unsigned short qs[8][16][64];    // per-wave Q tile
    __shared__ float ks[64];

    const int tid  = threadIdx.x;
    const int wave = tid >> 5;
    const int lane = tid & 31;
    const int lr   = lane & 15;
    const bool hi  = lane >= 16;
    const int g  = blockIdx.y;
    const int s0 = blockIdx.x * 128;
    const int bb = g >> 4, hh = g & 15;

    if (tid < 64) ks[tid] = Ksum[(size_t)g * 64 + tid];
    #pragma unroll
    for (int j = 0; j < 16; ++j) {              // KV -> bf16, transposed
        int e = tid + j * 256;                  // contiguous global read
        KVt[e & 63][e >> 6] = f2bf(KV[(size_t)g * 4096 + e]);
    }
    #pragma unroll
    for (int j = 0; j < 4; ++j) {               // wave's 16x64 Q tile
        int c = lane + j * 32;
        int row = c >> 3, col = (c & 7) * 8;
        *(uint4*)&qs[wave][row][col] =
            *(const uint4*)(Qm + ((size_t)g * SEQ + s0 + wave * 16 + row) * HDIM + col);
    }
    __syncthreads();

    // denom for this lane's A-fragment row (lanes l and l+16 share row lr)
    float denom = 0.0f;
    for (int d = 0; d < 64; ++d) denom += bf2f(qs[wave][lr][d]) * ks[d];
    float scale = 1.0f / denom;
    if (scale != scale) scale = 0.0f;                           // nan -> 0
    scale = fminf(fmaxf(scale, -3.4028235e38f), 3.4028235e38f); // inf -> max

    FragAB afr[2];
    #pragma unroll
    for (int kx = 0; kx < 2; ++kx) {            // scaled A frags, K=d
        const int base = kx * 32 + (hi ? 8 : 0);
        FragAB t;
        t.q[0] = *(const uint4*)&qs[wave][lr][base];
        t.q[1] = *(const uint4*)&qs[wave][lr][base + 16];
        #pragma unroll
        for (int e = 0; e < 16; ++e) afr[kx].h[e] = f2bf(bf2f(t.h[e]) * scale);
    }

    FragC acc[4];
    #pragma unroll
    for (int i = 0; i < 4; ++i)
        #pragma unroll
        for (int r = 0; r < 8; ++r) acc[i].f[r] = 0.0f;

    #pragma unroll
    for (int kx = 0; kx < 2; ++kx) {
        FragAB b[4];
        #pragma unroll
        for (int vt = 0; vt < 4; ++vt) {
            const int bc = kx * 32 + (hi ? 8 : 0);
            b[vt].q[0] = *(const uint4*)&KVt[vt * 16 + lr][bc];
            b[vt].q[1] = *(const uint4*)&KVt[vt * 16 + lr][bc + 16];
        }
        #pragma unroll
        for (int vt = 0; vt < 4; ++vt)
            acc[vt].v = __builtin_amdgcn_wmma_f32_16x16x32_bf16(
                false, afr[kx].v, false, b[vt].v, (short)0, acc[vt].v, false, false);
    }

    #pragma unroll
    for (int vt = 0; vt < 4; ++vt) {
        #pragma unroll
        for (int r = 0; r < 8; ++r) {
            const int row = hi ? r + 8 : r;
            const int s = s0 + wave * 16 + row;
            const int v = vt * 16 + lr;
            Attn[((size_t)(bb * SEQ + s)) * HID + hh * 64 + v] = f2bf(acc[vt].f[r]);
        }
    }
}

// ---------------------------------------------------------------------------
extern "C" void kernel_launch(void* const* d_in, const int* in_sizes, int n_in,
                              void* d_out, int out_size, void* d_ws, size_t ws_size,
                              hipStream_t stream)
{
    const float* X    = (const float*)d_in[0];
    const int*   mask = (const int*)  d_in[1];
    const float* gb   = (const float*)d_in[2];
    const float* Wq   = (const float*)d_in[3];
    const float* bq   = (const float*)d_in[4];
    const float* Wk   = (const float*)d_in[5];
    const float* bk   = (const float*)d_in[6];
    const float* Wv   = (const float*)d_in[7];
    const float* bv   = (const float*)d_in[8];
    const float* Wo   = (const float*)d_in[9];
    const float* bo   = (const float*)d_in[10];
    float* out = (float*)d_out;
    (void)in_sizes; (void)n_in; (void)out_size; (void)ws_size;

    char* ws = (char*)d_ws;
    size_t off = 0;
    auto alloc = [&](size_t bytes) -> char* {
        char* p = ws + off;
        off += (bytes + 255) & ~(size_t)255;
        return p;
    };
    unsigned short* Xb  = (unsigned short*)alloc((size_t)MTOT * HID * 2);
    unsigned short* Wqb = (unsigned short*)alloc((size_t)HID * HID * 2);
    unsigned short* Wkb = (unsigned short*)alloc((size_t)HID * HID * 2);
    unsigned short* Wvb = (unsigned short*)alloc((size_t)HID * HID * 2);
    unsigned short* Wob = (unsigned short*)alloc((size_t)HID * HID * 2);
    unsigned short* Qm  = (unsigned short*)alloc((size_t)MTOT * HID * 2);
    unsigned short* Km  = (unsigned short*)alloc((size_t)MTOT * HID * 2);
    unsigned short* Vm  = (unsigned short*)alloc((size_t)MTOT * HID * 2);
    unsigned short* At  = (unsigned short*)alloc((size_t)MTOT * HID * 2);
    float* KV   = (float*)alloc((size_t)64 * 64 * 64 * 4);   // 1 MB (256-aligned)
    float* Ksum = (float*)alloc((size_t)64 * 64 * 4);        // contiguous after KV

    // 1) fp32 -> bf16
    k_f32_to_bf16<<<2048, 256, 0, stream>>>(X,  Xb,  MTOT * HID);
    k_f32_to_bf16<<<512,  256, 0, stream>>>(Wq, Wqb, HID * HID);
    k_f32_to_bf16<<<512,  256, 0, stream>>>(Wk, Wkb, HID * HID);
    k_f32_to_bf16<<<512,  256, 0, stream>>>(Wv, Wvb, HID * HID);
    k_f32_to_bf16<<<512,  256, 0, stream>>>(Wo, Wob, HID * HID);

    // 2) zero KV + Ksum (contiguous)
    const int nz = 64 * 64 * 64 + 64 * 64;
    k_zero<<<(nz + 255) / 256, 256, 0, stream>>>(KV, nz);

    // 3) fused QKV projections + feature maps
    dim3 ggrid(HID / 128, MTOT / 128);
    k_gemm<<<ggrid, 256, 0, stream>>>(Xb, Wqb, bq, Qm, nullptr, gb, mask, 1);
    k_gemm<<<ggrid, 256, 0, stream>>>(Xb, Wkb, bk, Km, nullptr, gb, mask, 2);
    k_gemm<<<ggrid, 256, 0, stream>>>(Xb, Wvb, bv, Vm, nullptr, gb, mask, 0);

    // 4) kv_context + k_sum
    k_kv<<<dim3(16, BATCH * NHEADS), 256, 0, stream>>>(Km, Vm, KV, Ksum);

    // 5) normalized attention output (bf16, row-major [B*S, H])
    k_attn<<<dim3(SEQ / 128, BATCH * NHEADS), 256, 0, stream>>>(Qm, KV, Ksum, At);

    // 6) output projection -> fp32 d_out
    k_gemm<<<ggrid, 256, 0, stream>>>(At, Wob, bo, nullptr, out, nullptr, nullptr, 3);
}
